// SSD_58488864637444
// MI455X (gfx1250) — matrix operand order repaired and checked by reading, not compiled, not run on hardware
//
#include <hip/hip_runtime.h>
#include <hip/hip_bf16.h>

typedef __attribute__((ext_vector_type(16))) _Float16 v16h;
typedef __attribute__((ext_vector_type(8)))  _Float16 v8h;
typedef __attribute__((ext_vector_type(8)))  float    v8f;

#define BATCH 16
#define NCLS  21
#define NBOX  8732

// ---------------- fp32 -> fp16 transposed weight conversion ----------------
// in:  HWIO flat = [k][co] with k=(kh*KW+kw)*Ci+ci
// out: [co][k]  (so a WMMA B-fragment lane reads 16 contiguous halves)
__global__ void k_cvt_f16_t(const float* __restrict__ in, _Float16* __restrict__ out,
                            int K, int Co) {
  int i = blockIdx.x * blockDim.x + threadIdx.x;
  if (i < K * Co) {
    int k = i / Co, co = i - k * Co;        // coalesced read over co
    out[(long)co * K + k] = (_Float16)in[i];
  }
}

// Shared helper: build A fragment (16x32 f16, contiguous k = 16*hi + i lane layout) from NHWC f32.
__device__ __forceinline__ v16h load_afrag(const float* __restrict__ abase,
                                           bool aval, bool full, int ci0, int Ci, int hi) {
  v16h afrag;
  if (aval && full) {
    const float4 f0 = *(const float4*)(abase + ci0 + hi * 16);
    const float4 f1 = *(const float4*)(abase + ci0 + hi * 16 + 4);
    const float4 f2 = *(const float4*)(abase + ci0 + hi * 16 + 8);
    const float4 f3 = *(const float4*)(abase + ci0 + hi * 16 + 12);
    afrag[0]  = (_Float16)f0.x; afrag[1]  = (_Float16)f0.y;
    afrag[2]  = (_Float16)f0.z; afrag[3]  = (_Float16)f0.w;
    afrag[4]  = (_Float16)f1.x; afrag[5]  = (_Float16)f1.y;
    afrag[6]  = (_Float16)f1.z; afrag[7]  = (_Float16)f1.w;
    afrag[8]  = (_Float16)f2.x; afrag[9]  = (_Float16)f2.y;
    afrag[10] = (_Float16)f2.z; afrag[11] = (_Float16)f2.w;
    afrag[12] = (_Float16)f3.x; afrag[13] = (_Float16)f3.y;
    afrag[14] = (_Float16)f3.z; afrag[15] = (_Float16)f3.w;
  } else {
#pragma unroll
    for (int i = 0; i < 16; ++i) {
      const int kloc = hi * 16 + i;
      const int ci   = ci0 + kloc;
      float v = 0.0f;
      if (aval && ci < Ci) v = abase[ci];
      afrag[i] = (_Float16)v;
    }
  }
  return afrag;
}

// ---------------- 4-wave implicit-GEMM conv: 64M x 64N per block ----------------
// Each wave owns 16 M-rows; all 4 waves share one 64-col B chunk staged in LDS.
__global__ __launch_bounds__(128) void k_conv_wmma4(
    const float* __restrict__ in, const _Float16* __restrict__ wT,
    const float* __restrict__ bias, float* __restrict__ out,
    int H, int W, int Ci, int Co, int KH, int KW,
    int stride, int pad, int dil, int Ho, int Wo,
    long outBatchStride, long outBase, int relu)
{
  __shared__ _Float16 smem[64 * 32];   // [col][kloc], 4 KB
  const int tid  = threadIdx.x;
  const int wave = tid >> 5;
  const int lane = tid & 31;
  const int row  = lane & 15;
  const int hi   = lane >> 4;
  const int tileM = blockIdx.x;
  const int tileN = blockIdx.y;
  const int M = BATCH * Ho * Wo;
  const int K = KH * KW * Ci;
  const long obs = (outBatchStride < 0) ? (long)Ho * Wo * Co : outBatchStride;

  // A-row pixel decode (one div per thread)
  const int am = tileM * 64 + wave * 16 + row;
  const bool arow_ok = (am < M);
  int ab = 0, aoh = 0, aow = 0;
  if (arow_ok) { ab = am / (Ho * Wo); int pix = am - ab * Ho * Wo; aoh = pix / Wo; aow = pix - aoh * Wo; }
  const int ihs = aoh * stride - pad;
  const int iws = aow * stride - pad;

  // B staging ids: 128 threads cover 64 cols x 2 half-chunks of 16 halves
  const int scol  = tid & 63;
  const int shsel = tid >> 6;
  const int sncol = tileN * 64 + scol;
  const _Float16* swp = wT + (size_t)(sncol < Co ? sncol : 0) * K;
  _Float16* sdst = smem + scol * 32 + shsel * 16;

  int ncol[4]; bool nok[4];
#pragma unroll
  for (int t = 0; t < 4; ++t) {
    ncol[t] = tileN * 64 + t * 16 + row;
    nok[t]  = (ncol[t] < Co);
  }

  v8f acc[4] = {};

  for (int kh = 0; kh < KH; ++kh) {
    const int ih = ihs + kh * dil;
    for (int kw = 0; kw < KW; ++kw) {
      const int iw = iws + kw * dil;
      const bool aval = arow_ok && (ih >= 0) && (ih < H) && (iw >= 0) && (iw < W);
      const float* abase = in + (((long)ab * H + ih) * W + iw) * Ci;
      const int kflat = (kh * KW + kw) * Ci;

      for (int ci0 = 0; ci0 < Ci; ci0 += 32) {
        const bool full = (ci0 + 32 <= Ci);

        // ---- cooperative B-chunk stage: 64 cols x 32 kloc halves ----
        if (sncol < Co && full) {
          const _Float16* p = swp + kflat + ci0 + shsel * 16;
          const v8h s0 = *(const v8h*)p;
          const v8h s1 = *(const v8h*)(p + 8);
          *(v8h*)sdst       = s0;
          *(v8h*)(sdst + 8) = s1;
        } else {
#pragma unroll
          for (int j = 0; j < 16; ++j) {
            const int ci = ci0 + shsel * 16 + j;
            sdst[j] = (sncol < Co && ci < Ci) ? swp[kflat + ci] : (_Float16)0.0f;
          }
        }
        __syncthreads();

        if (aval && full && ci0 + 64 <= Ci)
          __builtin_prefetch((const void*)(abase + ci0 + 32), 0, 1);
        const v16h afrag = load_afrag(abase, aval, full, ci0, Ci, hi);

#pragma unroll
        for (int t = 0; t < 4; ++t) {
          const _Float16* bp = smem + (t * 16 + row) * 32 + hi * 16;
          const v8h b0 = *(const v8h*)bp;          // ds_load_b128
          const v8h b1 = *(const v8h*)(bp + 8);    // ds_load_b128
          v16h bfrag;
#pragma unroll
          for (int i = 0; i < 8; ++i) { bfrag[i] = b0[i]; bfrag[i + 8] = b1[i]; }
          acc[t] = __builtin_amdgcn_wmma_f32_16x16x32_f16(false, afrag, false, bfrag,
                                                          (short)0, acc[t], false, false);
        }
        __syncthreads();   // protect smem before next chunk's stage
      }
    }
  }

  // ---- epilogue: hoist M decode out of the t loop ----
  int  bbr[8], pixr[8]; bool mok[8];
#pragma unroll
  for (int r = 0; r < 8; ++r) {
    const int mm = tileM * 64 + wave * 16 + hi * 8 + r;
    mok[r] = (mm < M);
    int bb = 0, pix = 0;
    if (mok[r]) { bb = mm / (Ho * Wo); pix = mm - bb * Ho * Wo; }
    bbr[r] = bb; pixr[r] = pix;
  }
#pragma unroll
  for (int t = 0; t < 4; ++t) {
    if (!nok[t]) continue;
    const float bv = bias[ncol[t]];
#pragma unroll
    for (int r = 0; r < 8; ++r) {
      if (mok[r]) {
        float v = acc[t][r] + bv;
        if (relu) v = fmaxf(v, 0.0f);
        out[(long)bbr[r] * obs + outBase + (long)pixr[r] * Co + ncol[t]] = v;
      }
    }
  }
}

// ---------------- single-wave 16x16 conv (small Co heads) ----------------
__global__ __launch_bounds__(32) void k_conv_wmma1(
    const float* __restrict__ in, const _Float16* __restrict__ wT,
    const float* __restrict__ bias, float* __restrict__ out,
    int H, int W, int Ci, int Co, int KH, int KW,
    int stride, int pad, int dil, int Ho, int Wo,
    long outBatchStride, long outBase, int relu)
{
  const int lane = threadIdx.x & 31;
  const int row  = lane & 15;
  const int hi   = lane >> 4;
  const int tileM = blockIdx.x;
  const int tileN = blockIdx.y;
  const int M = BATCH * Ho * Wo;
  const int K = KH * KW * Ci;
  const long obs = (outBatchStride < 0) ? (long)Ho * Wo * Co : outBatchStride;

  const int am = tileM * 16 + row;
  const bool arow_ok = (am < M);
  int ab = 0, aoh = 0, aow = 0;
  if (arow_ok) { ab = am / (Ho * Wo); int pix = am - ab * Ho * Wo; aoh = pix / Wo; aow = pix - aoh * Wo; }
  const int ihs = aoh * stride - pad;
  const int iws = aow * stride - pad;

  const int  ncol    = tileN * 16 + row;
  const bool ncol_ok = (ncol < Co);
  const _Float16* wcol = wT + (size_t)(ncol_ok ? ncol : 0) * K;

  v8f acc = {};
  for (int kh = 0; kh < KH; ++kh) {
    const int ih = ihs + kh * dil;
    for (int kw = 0; kw < KW; ++kw) {
      const int iw = iws + kw * dil;
      const bool aval = arow_ok && (ih >= 0) && (ih < H) && (iw >= 0) && (iw < W);
      const float* abase = in + (((long)ab * H + ih) * W + iw) * Ci;
      const int kflat = (kh * KW + kw) * Ci;
      for (int ci0 = 0; ci0 < Ci; ci0 += 32) {
        const bool full = (ci0 + 32 <= Ci);
        const v16h afrag = load_afrag(abase, aval, full, ci0, Ci, hi);
        v16h bfrag;
        if (ncol_ok && full) {
          const _Float16* bp = wcol + kflat + ci0 + hi * 16;
          const v8h b0 = *(const v8h*)bp;
          const v8h b1 = *(const v8h*)(bp + 8);
#pragma unroll
          for (int i = 0; i < 8; ++i) { bfrag[i] = b0[i]; bfrag[i + 8] = b1[i]; }
        } else {
#pragma unroll
          for (int i = 0; i < 16; ++i) {
            const int ci = ci0 + hi * 16 + i;
            bfrag[i] = (ncol_ok && ci < Ci) ? wcol[kflat + ci] : (_Float16)0.0f;
          }
        }
        acc = __builtin_amdgcn_wmma_f32_16x16x32_f16(false, afrag, false, bfrag,
                                                     (short)0, acc, false, false);
      }
    }
  }

  if (ncol_ok) {
    const float bv = bias[ncol];
#pragma unroll
    for (int r = 0; r < 8; ++r) {
      const int mm = tileM * 16 + hi * 8 + r;
      if (mm < M) {
        float v = acc[r] + bv;
        if (relu) v = fmaxf(v, 0.0f);
        const int bb  = mm / (Ho * Wo);
        const int pix = mm - bb * Ho * Wo;
        out[(long)bb * obs + outBase + (long)pix * Co + ncol] = v;
      }
    }
  }
}

// ---------------- max pool (-inf padding, torch semantics) ----------------
__global__ void k_maxpool(const float* __restrict__ in, float* __restrict__ out,
                          int H, int W, int C, int k, int s, int padlo,
                          int Ho, int Wo)
{
  long total = (long)BATCH * Ho * Wo * C;
  long i = (long)blockIdx.x * blockDim.x + threadIdx.x;
  if (i >= total) return;
  int c  = (int)(i % C); long t = i / C;
  int ow = (int)(t % Wo); t /= Wo;
  int oh = (int)(t % Ho);
  int b  = (int)(t / Ho);
  float m = -__builtin_inff();
  for (int dy = 0; dy < k; ++dy) {
    int ih = oh * s - padlo + dy;
    if (ih < 0 || ih >= H) continue;
    for (int dx = 0; dx < k; ++dx) {
      int iw = ow * s - padlo + dx;
      if (iw < 0 || iw >= W) continue;
      m = fmaxf(m, in[(((long)b * H + ih) * W + iw) * C + c]);
    }
  }
  out[i] = m;
}

// ---------------- L2Norm over channels ----------------
__global__ void k_l2norm(const float* __restrict__ in, const float* __restrict__ scale,
                         float* __restrict__ out, int pixels, int C)
{
  int p = blockIdx.x * blockDim.x + threadIdx.x;
  if (p >= pixels) return;
  const float* x = in + (long)p * C;
  float s = 0.0f;
  for (int c = 0; c < C; ++c) { float v = x[c]; s += v * v; }
  float r = rsqrtf(s + 1e-10f);
  float* o = out + (long)p * C;
  for (int c = 0; c < C; ++c) o[c] = x[c] * scale[c] * r;
}

__global__ void k_copy(const float* __restrict__ in, float* __restrict__ out, int n) {
  int i = blockIdx.x * blockDim.x + threadIdx.x;
  if (i < n) out[i] = in[i];
}

// ---------------- host orchestration ----------------
extern "C" void kernel_launch(void* const* d_in, const int* in_sizes, int n_in,
                              void* d_out, int out_size, void* d_ws, size_t ws_size,
                              hipStream_t stream)
{
  (void)in_sizes; (void)n_in; (void)out_size; (void)ws_size;

  static const int VGG_K[15][4] = {
    {3,3,3,64},{3,3,64,64},{3,3,64,128},{3,3,128,128},{3,3,128,256},
    {3,3,256,256},{3,3,256,256},{3,3,256,512},{3,3,512,512},{3,3,512,512},
    {3,3,512,512},{3,3,512,512},{3,3,512,512},{3,3,512,1024},{1,1,1024,1024}};
  static const int EX_K[8][4] = {
    {1,1,1024,256},{3,3,256,512},{1,1,512,128},{3,3,128,256},
    {1,1,256,128},{3,3,128,256},{1,1,256,128},{3,3,128,256}};
  static const int EX_SP[8][2] = {{1,0},{2,1},{1,0},{2,1},{1,0},{1,0},{1,0},{1,0}};
  static const int SRC_CH[6]  = {512,1024,512,256,256,256};
  static const int ANC[6]     = {4,6,6,6,4,4};
  static const int SRC_OFF[6] = {0,5776,7942,8542,8692,8728};

  // d_in layout (setup_inputs order, lists flattened):
  // 0:x  1-15:vgg_w  16-30:vgg_b  31-38:extra_w  39-46:extra_b
  // 47-52:loc_w 53-58:loc_b 59-64:conf_w 65-70:conf_b 71:l2_scale 72:dbox

  _Float16* wbase = (_Float16*)d_ws;
  size_t woff = 0;
  auto walloc = [&](size_t elems) -> _Float16* {
    _Float16* p = wbase + woff;
    woff += (elems + 127) & ~(size_t)127;
    return p;
  };
  auto cvt = [&](const void* src, _Float16* dst, int K, int Co) {
    int nn = K * Co;
    k_cvt_f16_t<<<dim3((nn + 255) / 256), dim3(256), 0, stream>>>(
        (const float*)src, dst, K, Co);
  };

  _Float16 *wv[15], *we[8], *wl[6], *wc[6];
  for (int i = 0; i < 15; ++i) {
    int K = VGG_K[i][0] * VGG_K[i][1] * VGG_K[i][2], Co = VGG_K[i][3];
    wv[i] = walloc((size_t)K * Co); cvt(d_in[1 + i], wv[i], K, Co);
  }
  for (int i = 0; i < 8; ++i) {
    int K = EX_K[i][0] * EX_K[i][1] * EX_K[i][2], Co = EX_K[i][3];
    we[i] = walloc((size_t)K * Co); cvt(d_in[31 + i], we[i], K, Co);
  }
  for (int i = 0; i < 6; ++i) {
    int K = 9 * SRC_CH[i], Co = ANC[i] * 4;
    wl[i] = walloc((size_t)K * Co); cvt(d_in[47 + i], wl[i], K, Co);
  }
  for (int i = 0; i < 6; ++i) {
    int K = 9 * SRC_CH[i], Co = ANC[i] * NCLS;
    wc[i] = walloc((size_t)K * Co); cvt(d_in[59 + i], wc[i], K, Co);
  }

  char*  base   = (char*)d_ws;
  size_t actOff = (woff * sizeof(_Float16) + 255) & ~(size_t)255;
  const size_t ABUF = (size_t)BATCH * 300 * 300 * 64;  // floats (largest activation)
  float* bufA = (float*)(base + actOff);
  float* bufB = bufA + ABUF;
  float* bufS = bufB + ABUF;                           // L2Norm output buffer

  auto conv = [&](const float* in, int H, int W, int Ci, const _Float16* w,
                  const void* bias, int Co, int KH, int KW, int s, int p, int d,
                  float* out, long obs, long obase, int relu, int& Ho, int& Wo) {
    Ho = (H + 2 * p - d * (KH - 1) - 1) / s + 1;
    Wo = (W + 2 * p - d * (KW - 1) - 1) / s + 1;
    int M = BATCH * Ho * Wo;
    if (Co >= 64) {
      dim3 grid((M + 63) / 64, (Co + 63) / 64);
      k_conv_wmma4<<<grid, dim3(128), 0, stream>>>(in, w, (const float*)bias, out,
          H, W, Ci, Co, KH, KW, s, p, d, Ho, Wo, obs, obase, relu);
    } else {
      dim3 grid((M + 15) / 16, (Co + 15) / 16);
      k_conv_wmma1<<<grid, dim3(32), 0, stream>>>(in, w, (const float*)bias, out,
          H, W, Ci, Co, KH, KW, s, p, d, Ho, Wo, obs, obase, relu);
    }
  };
  auto pool = [&](const float* in, int H, int W, int C, int k, int s, int padlo,
                  int Ho, int Wo, float* out) {
    long total = (long)BATCH * Ho * Wo * C;
    k_maxpool<<<dim3((unsigned)((total + 255) / 256)), dim3(256), 0, stream>>>(
        in, out, H, W, C, k, s, padlo, Ho, Wo);
  };

  float* outF = (float*)d_out;
  const long LOC_TOT = (long)BATCH * NBOX * 4;
  auto heads = [&](const float* src, int H, int W, int s) {
    int ho, wo;
    int ch = SRC_CH[s], a = ANC[s];
    conv(src, H, W, ch, wl[s], d_in[53 + s], a * 4,    3, 3, 1, 1, 1,
         outF,           (long)NBOX * 4,    (long)SRC_OFF[s] * 4,    0, ho, wo);
    conv(src, H, W, ch, wc[s], d_in[65 + s], a * NCLS, 3, 3, 1, 1, 1,
         outF + LOC_TOT, (long)NBOX * NCLS, (long)SRC_OFF[s] * NCLS, 0, ho, wo);
  };

  int ho, wo;
  const float* X = (const float*)d_in[0];

  // ---- VGG trunk ----
  conv(X,    300, 300,   3, wv[0],  d_in[16],  64, 3, 3, 1, 1, 1, bufA, -1, 0, 1, ho, wo);
  conv(bufA, 300, 300,  64, wv[1],  d_in[17],  64, 3, 3, 1, 1, 1, bufB, -1, 0, 1, ho, wo);
  pool(bufB, 300, 300,  64, 2, 2, 0, 150, 150, bufA);
  conv(bufA, 150, 150,  64, wv[2],  d_in[18], 128, 3, 3, 1, 1, 1, bufB, -1, 0, 1, ho, wo);
  conv(bufB, 150, 150, 128, wv[3],  d_in[19], 128, 3, 3, 1, 1, 1, bufA, -1, 0, 1, ho, wo);
  pool(bufA, 150, 150, 128, 2, 2, 0, 75, 75, bufB);
  conv(bufB,  75,  75, 128, wv[4],  d_in[20], 256, 3, 3, 1, 1, 1, bufA, -1, 0, 1, ho, wo);
  conv(bufA,  75,  75, 256, wv[5],  d_in[21], 256, 3, 3, 1, 1, 1, bufB, -1, 0, 1, ho, wo);
  conv(bufB,  75,  75, 256, wv[6],  d_in[22], 256, 3, 3, 1, 1, 1, bufA, -1, 0, 1, ho, wo);
  pool(bufA,  75,  75, 256, 2, 2, 0, 38, 38, bufB);   // ceil_mode: pad hi=1 -> Ho=38
  conv(bufB,  38,  38, 256, wv[7],  d_in[23], 512, 3, 3, 1, 1, 1, bufA, -1, 0, 1, ho, wo);
  conv(bufA,  38,  38, 512, wv[8],  d_in[24], 512, 3, 3, 1, 1, 1, bufB, -1, 0, 1, ho, wo);
  conv(bufB,  38,  38, 512, wv[9],  d_in[25], 512, 3, 3, 1, 1, 1, bufA, -1, 0, 1, ho, wo); // conv4_3

  // L2Norm -> source 0, heads
  {
    int pixels = BATCH * 38 * 38;
    k_l2norm<<<dim3((pixels + 255) / 256), dim3(256), 0, stream>>>(
        bufA, (const float*)d_in[71], bufS, pixels, 512);
  }
  heads(bufS, 38, 38, 0);

  pool(bufA, 38, 38, 512, 2, 2, 0, 19, 19, bufB);
  conv(bufB, 19, 19, 512,  wv[10], d_in[26],  512, 3, 3, 1, 1, 1, bufA, -1, 0, 1, ho, wo);
  conv(bufA, 19, 19, 512,  wv[11], d_in[27],  512, 3, 3, 1, 1, 1, bufB, -1, 0, 1, ho, wo);
  conv(bufB, 19, 19, 512,  wv[12], d_in[28],  512, 3, 3, 1, 1, 1, bufA, -1, 0, 1, ho, wo);
  pool(bufA, 19, 19, 512, 3, 1, 1, 19, 19, bufB);     // pool5
  conv(bufB, 19, 19, 512,  wv[13], d_in[29], 1024, 3, 3, 1, 6, 6, bufA, -1, 0, 1, ho, wo); // conv6 dilated
  conv(bufA, 19, 19, 1024, wv[14], d_in[30], 1024, 1, 1, 1, 0, 1, bufB, -1, 0, 1, ho, wo); // conv7
  heads(bufB, 19, 19, 1);

  // ---- extras (sources after every odd conv) ----
  {
    const float* cur = bufB;
    float* dst = bufA;
    int eh = 19, ew = 19, srcIdx = 2;
    for (int i = 0; i < 8; ++i) {
      conv(cur, eh, ew, EX_K[i][2], we[i], d_in[39 + i], EX_K[i][3],
           EX_K[i][0], EX_K[i][1], EX_SP[i][0], EX_SP[i][1], 1,
           dst, -1, 0, 1, ho, wo);
      cur = dst; dst = (dst == bufA) ? bufB : bufA;
      eh = ho; ew = wo;
      if (i & 1) heads(cur, eh, ew, srcIdx++);
    }
  }

  // ---- dbox passthrough ----
  {
    int n = NBOX * 4;
    k_copy<<<dim3((n + 255) / 256), dim3(256), 0, stream>>>(
        (const float*)d_in[72], outF + (long)BATCH * NBOX * (4 + NCLS), n);
  }
}
